// SharpenedCosineSimilarity_34419867910708
// MI455X (gfx1250) — matrix operand order repaired and checked by reading, not compiled, not run on hardware
//
#include <hip/hip_runtime.h>
#include <hip/hip_bf16.h>
#include <math.h>

// ---------------- problem constants ----------------
#define NB   16          // batch
#define CIN  64          // input channels
#define HH   64
#define WW   64
#define VOUT 128         // output channels
#define LL   9           // 3x3 taps
#define KK   (CIN * LL)  // 576 reduction length
#define PIX  (NB * HH * WW)   // 65536
#define EPSV 1e-12f

// LDS padded strides (in bf16 elements): the 16 distinct fragment-row starts
// hit 16 distinct 4-bank spans for ds_load_b128 reads.
#define BP_STRIDE 584    // patch rows: start = r*292 dw, 292%64=36 -> distinct mult-of-4 spans
#define WT_STRIDE 40     // weight rows: start = r*20 dw -> distinct mult-of-4 spans

#define PTILE 64         // pixels per block (full W row)

typedef __attribute__((ext_vector_type(16))) __bf16 v16bf;
typedef __attribute__((ext_vector_type(8)))  float  v8f;
typedef int v4i_t __attribute__((vector_size(16)));   // matches async-LDS builtin param

struct __attribute__((aligned(16))) U4 { unsigned int x, y, z, w; };
struct __attribute__((aligned(16))) U8 { U4 lo, hi; };   // 32B == v16bf

#if defined(__gfx1250__) && \
    __has_builtin(__builtin_amdgcn_global_load_async_to_lds_b128) && \
    __has_builtin(__builtin_amdgcn_s_wait_asynccnt)
#define USE_ASYNC_LDS 1
#define GAS1(p) ((__attribute__((address_space(1))) v4i_t*)(p))
#define LAS3(p) ((__attribute__((address_space(3))) v4i_t*)(p))
#else
#define USE_ASYNC_LDS 0
#endif

__device__ __forceinline__ unsigned short f2bf(float f) {
    unsigned int u = __float_as_uint(f);
    unsigned int r = (u + 0x7FFFu + ((u >> 16) & 1u)) >> 16;   // RNE
    return (unsigned short)r;
}

// ---------------- kernel 1: normalize weights, compute expo ----------------
__global__ __launch_bounds__(256) void scs_prep_weights(
    const float* __restrict__ w,      // [128,576]
    const float* __restrict__ p,      // [128]
    const float* __restrict__ q,      // [1]
    unsigned short* __restrict__ wq,  // [128,576] bf16 out
    float* __restrict__ expo)         // [128] out
{
    __shared__ float red[256];
    const int v = blockIdx.x;
    const int t = threadIdx.x;
    const float* wr = w + v * KK;

    float acc = 0.f;
    for (int k = t; k < KK; k += 256) { float a = wr[k]; acc += a * a; }
    red[t] = acc;
    __syncthreads();
    for (int s = 128; s > 0; s >>= 1) {
        if (t < s) red[t] += red[t + s];
        __syncthreads();
    }
    const float qn    = q[0] * 0.01f;
    const float scale = qn * qn;
    const float inv   = 1.0f / (sqrtf(red[0] + EPSV) + scale);

    unsigned short* dst = wq + v * KK;
    for (int k = t; k < KK; k += 256) dst[k] = f2bf(wr[k] * inv);
    if (t == 0) { float e = p[v] * 0.1f; expo[v] = e * e; }
}

// ---------------- kernel 2: per-pixel sum over channels of x^2 ----------------
__global__ __launch_bounds__(256) void scs_sumsq(
    const float* __restrict__ x,   // [16,64,64,64]
    float* __restrict__ s)         // [65536]
{
    const int pI = blockIdx.x * 256 + threadIdx.x;
    const int n  = pI >> 12;
    const int hw = pI & 4095;
    const float* xb = x + ((size_t)n * CIN << 12) + hw;
    float acc = 0.f;
    #pragma unroll 4
    for (int c = 0; c < CIN; ++c) { float a = xb[(size_t)c << 12]; acc += a * a; }
    s[pI] = acc;
}

// ---------------- kernel 3: 3x3 window -> x_norm ----------------
__global__ __launch_bounds__(256) void scs_xnorm(
    const float* __restrict__ s,   // [65536]
    const float* __restrict__ q,   // [1]
    float* __restrict__ xn)        // [65536]
{
    const int pI = blockIdx.x * 256 + threadIdx.x;
    const int n  = pI >> 12;
    const int h  = (pI >> 6) & 63;
    const int w  = pI & 63;
    const float* sb = s + (n << 12);
    float acc = 0.f;
    #pragma unroll
    for (int a = -1; a <= 1; ++a) {
        int hh = h + a;
        if (hh < 0 || hh >= HH) continue;
        #pragma unroll
        for (int b = -1; b <= 1; ++b) {
            int ww = w + b;
            if (ww < 0 || ww >= WW) continue;
            acc += sb[hh * WW + ww];
        }
    }
    const float qn = q[0] * 0.01f;
    xn[pI] = sqrtf(acc + EPSV) + qn * qn;
}

// ---------------- kernel 4: implicit-GEMM conv + SCS epilogue ----------------
// Block = 256 threads (8 waves). Tile: 128 channels x 64 pixels (one W row).
// Per wave, per K-chunk of 32: 1 A fragment (16 ch x 32 K), 4 B fragments
// (32 K x 16 px each) -> 4x v_wmma_f32_16x16x32_bf16.
__global__ __launch_bounds__(256) void scs_main(
    const float* __restrict__ x,            // [16,64,64,64]
    const unsigned short* __restrict__ wq,  // [128,576] bf16 normalized
    const float* __restrict__ expo,         // [128]
    const float* __restrict__ xnorm,        // [65536]
    float* __restrict__ out)                // [16,128,64,64]
{
    __shared__ __attribute__((aligned(16))) unsigned short Bp[PTILE * BP_STRIDE]; // patch tile
    __shared__ __attribute__((aligned(16))) unsigned short Wt[VOUT * WT_STRIDE];  // weight K-chunk
    __shared__ float exL[VOUT];
    __shared__ float xnL[PTILE];

    const int pb = blockIdx.x * PTILE;      // first pixel of tile (w0 == 0)
    const int n  = pb >> 12;
    const int h  = (pb >> 6) & 63;

    const int t    = threadIdx.x;
    const int wv   = t >> 5;                // wave id -> channel tile 16*wv
    const int lane = t & 31;
    const int mrow = lane & 15;
    const int kg   = lane >> 4;             // K half-group selector

    // ---- stage epilogue constants ----
    if (t < VOUT)  exL[t] = expo[t];
    if (t < PTILE) xnL[t] = xnorm[pb + t];

    // ---- build patch tile in LDS (64 px x 576 K, bf16) ----
    #pragma unroll
    for (int it = 0; it < 16; ++it) {
        const int idx = it * 256 + t;       // 0..4095
        const int pix = idx & 63;           // == w
        const int c   = idx >> 6;
        const float* xb = x + (((size_t)(n * CIN + c)) << 12);
        unsigned short* dst = &Bp[pix * BP_STRIDE + c * LL];
        #pragma unroll
        for (int a = 0; a < 3; ++a) {
            const int hh = h + a - 1;
            const bool hok = (hh >= 0) && (hh < HH);
            #pragma unroll
            for (int b = 0; b < 3; ++b) {
                const int ww = pix + b - 1;
                float v = (hok && ww >= 0 && ww < WW) ? xb[hh * WW + ww] : 0.f;
                dst[a * 3 + b] = f2bf(v);
            }
        }
    }

    v8f acc[4] = {};    // pixel groups 0..3 (pixels 16*pg .. 16*pg+15)

    const int arow = (wv * 16 + mrow) * WT_STRIDE;

    for (int kc = 0; kc < KK / 32; ++kc) {
        __syncthreads();   // prior chunk fully consumed by all waves
        // ---- cooperative stage of weight K-chunk: thread -> (v = t/2, half = t&1) ----
        {
            const int vrow = t >> 1;
            const int half = t & 1;
            unsigned short* gsrc = (unsigned short*)(wq + vrow * KK + kc * 32 + half * 16);
            unsigned short* ldst = &Wt[vrow * WT_STRIDE + half * 16];
#if USE_ASYNC_LDS
            __builtin_amdgcn_global_load_async_to_lds_b128(GAS1(gsrc),     LAS3(ldst),     0, 0);
            __builtin_amdgcn_global_load_async_to_lds_b128(GAS1(gsrc + 8), LAS3(ldst + 8), 0, 0);
            __builtin_amdgcn_s_wait_asynccnt(0);
#else
            const U4* src = (const U4*)gsrc;
            U4* dst = (U4*)ldst;
            dst[0] = src[0];
            dst[1] = src[1];
#endif
        }
        __syncthreads();

        // A fragment: lane holds channel row (mrow), contiguous K {kg*16..+16}
        U8 af;
        af.lo = *(const U4*)&Wt[arow + kg * 16];
        af.hi = *(const U4*)&Wt[arow + kg * 16 + 8];
        const v16bf av = __builtin_bit_cast(v16bf, af);

        // B fragments: lane holds pixel column, contiguous K {kg*16..+16}
        const int kbase = kc * 32 + kg * 16;
        #pragma unroll
        for (int pg = 0; pg < 4; ++pg) {
            const int brow = (pg * 16 + mrow) * BP_STRIDE;
            U8 bf;
            bf.lo = *(const U4*)&Bp[brow + kbase];
            bf.hi = *(const U4*)&Bp[brow + kbase + 8];
            const v16bf bv = __builtin_bit_cast(v16bf, bf);
            acc[pg] = __builtin_amdgcn_wmma_f32_16x16x32_bf16(
                false, av, false, bv, (short)0, acc[pg], false, false);
        }
    }

    // ---- epilogue: y/x_norm, sign * (|y|+eps)^expo, NCHW store ----
    // C/D layout: VGPR r -> lanes 0-15: M=r, N=lane; lanes 16-31: M=8+r, N=lane-16
    const int vbase = wv * 16 + (lane >> 4) * 8;
    #pragma unroll
    for (int pg = 0; pg < 4; ++pg) {
        const int pixl = pg * 16 + (lane & 15);    // == w coordinate
        const float inv = 1.0f / xnL[pixl];
        #pragma unroll
        for (int r = 0; r < 8; ++r) {
            const int vch = vbase + r;
            const float y  = acc[pg][r] * inv;
            const float e  = exL[vch];
            const float pw = powf(fabsf(y) + EPSV, e);
            const float sg = (y > 0.f) ? 1.f : ((y < 0.f) ? -1.f : 0.f);
            out[(((size_t)(n * VOUT + vch)) << 12) + h * WW + pixl] = sg * pw;
        }
    }
}

// ---------------- launcher ----------------
extern "C" void kernel_launch(void* const* d_in, const int* in_sizes, int n_in,
                              void* d_out, int out_size, void* d_ws, size_t ws_size,
                              hipStream_t stream) {
    const float* x  = (const float*)d_in[0];   // [16,64,64,64]
    const float* w  = (const float*)d_in[1];   // [128,64,9]
    const float* p  = (const float*)d_in[2];   // [128]
    const float* q  = (const float*)d_in[3];   // [1]
    float* out = (float*)d_out;

    // workspace layout
    char* ws = (char*)d_ws;
    unsigned short* wq   = (unsigned short*)ws;                       // 147456 B
    float*          exp_ = (float*)(ws + 147456);                     // 512 B
    float*          s    = (float*)(ws + 147968);                     // 262144 B
    float*          xn   = (float*)(ws + 147968 + 262144);            // 262144 B
    (void)in_sizes; (void)n_in; (void)out_size; (void)ws_size;

    scs_prep_weights<<<VOUT, 256, 0, stream>>>(w, p, q, wq, exp_);
    scs_sumsq<<<PIX / 256, 256, 0, stream>>>(x, s);
    scs_xnorm<<<PIX / 256, 256, 0, stream>>>(s, q, xn);
    scs_main<<<PIX / PTILE, 256, 0, stream>>>(x, wq, exp_, xn, out);
}
